// MLPConv_60507499266526
// MI455X (gfx1250) — matrix operand order repaired and checked by reading, not compiled, not run on hardware
//
#include <hip/hip_runtime.h>

// Problem constants
#define B_    32
#define HI    64
#define WI    64
#define C_    128
#define F_    256
#define HO    62
#define WO    62
#define NPOS  (HO * WO)          // 3844
#define NB    (NPOS * B_)        // 123008 rows total / out f-stride

// LDS strides (padded: row stride mod 64 dwords == 4 -> conflict-free frags)
#define XS_STRIDE 136            // 128 ch + 8 pad (bf16)  -> 272 B
#define WS_STRIDE 136            // 128 k  + 8 pad (bf16)  -> 272 B
#define H1_STRIDE 264            // 256 + 8 pad (bf16)     -> 528 B
#define H2_STRIDE 65             // 64 rows + 1 pad (f32)

// LDS layout (bytes)
#define XS_OFF  0
#define H1_OFF  (64 * XS_STRIDE * 2)                    // 17408
#define WS_OFF  (H1_OFF + 64 * H1_STRIDE * 2)           // 51200
#define SMEM_SZ (WS_OFF + 256 * WS_STRIDE * 2)          // 120832 (Ws/H2s union)

typedef __attribute__((ext_vector_type(16))) __bf16        bf16x16;
typedef __attribute__((ext_vector_type(8)))  float         f32x8;
typedef __attribute__((ext_vector_type(4)))  unsigned int  uint4v;

union Frag {
    uint4v  u[2];
    bf16x16 v;
};

__device__ __forceinline__ unsigned short f2bf(float f) {
    unsigned int u = __float_as_uint(f);
    u += 0x7FFFu + ((u >> 16) & 1u);     // round-to-nearest-even
    return (unsigned short)(u >> 16);
}

// ASYNCcnt-tracked DMA: 16 B global -> LDS, no VGPR round trip.
// VDST = wave-relative LDS byte offset (low 32 bits of a generic LDS pointer).
__device__ __forceinline__ void async_ld_b128(unsigned lds_off, const void* g) {
    asm volatile("global_load_async_to_lds_b128 %0, %1, off"
                 :: "v"(lds_off), "v"(g) : "memory");
}
__device__ __forceinline__ void wait_async0() {
    asm volatile("s_wait_asynccnt 0x0" ::: "memory");
}

// ---------------------------------------------------------------------------
// Prep: transpose + convert weights to bf16 in workspace.
//   w0c: [9][256][128]  w0c[p][n][c] = bf16( W0[(p*128+c)*256 + n] )
//   w1t: [256][256]     w1t[n][k]    = bf16( W1[k*256 + n] )
// ---------------------------------------------------------------------------
__global__ void prep_weights(const float* __restrict__ w0,
                             const float* __restrict__ w1,
                             unsigned short* __restrict__ w0c,
                             unsigned short* __restrict__ w1t) {
    const int i = blockIdx.x * blockDim.x + threadIdx.x;
    const int n0c = 9 * F_ * C_;                    // 294912
    if (i < n0c) {
        const int p = i / (F_ * C_);
        const int r = i % (F_ * C_);
        const int n = r / C_;
        const int c = r % C_;
        w0c[i] = f2bf(w0[(p * C_ + c) * F_ + n]);
    } else if (i < n0c + F_ * F_) {
        const int j = i - n0c;
        const int n = j / F_;
        const int k = j % F_;
        w1t[j] = f2bf(w1[k * F_ + n]);
    }
}

// ---------------------------------------------------------------------------
// Fused im2col + GEMM(1152->256) + ReLU + GEMM(256->256) + ReLU + permute-out.
// Block: 64 rows = (2 output positions) x (32 batches), all 256 columns.
// 8 waves, each computes 2 M-tiles x 4 N-tiles (8 accumulators).
// Layer-1 B operands are DMA'd to LDS with global_load_async_to_lds_b128.
// ---------------------------------------------------------------------------
__global__ __launch_bounds__(256)
void mlpconv_fused(const float* __restrict__ in,
                   const unsigned short* __restrict__ w0c_,
                   const float* __restrict__ b0,
                   const unsigned short* __restrict__ w1t_,
                   const float* __restrict__ b1,
                   float* __restrict__ out) {
    extern __shared__ char smem[];
    __bf16*         Xs   = (__bf16*)(smem + XS_OFF);
    __bf16*         H1s  = (__bf16*)(smem + H1_OFF);
    unsigned short* H1su = (unsigned short*)H1s;
    __bf16*         Ws   = (__bf16*)(smem + WS_OFF);   // layer-1 weights
    float*          H2s  = (float*)(smem + WS_OFF);    // union: dead after L1

    const int t     = threadIdx.x;
    const int lane  = t & 31;
    const int lrow  = lane & 15;
    const int lhalf = lane >> 4;
    const int wave  = t >> 5;
    const int mi2   = wave & 1;          // M pair: tiles {2*mi2, 2*mi2+1}
    const int q     = wave >> 1;         // N quad: tiles 4q..4q+3
    const int n0    = blockIdx.x * 2;    // first output position of this block

    const __bf16* w1t = (const __bf16*)w1t_;
    const unsigned ws_base = (unsigned)(size_t)(void*)Ws;  // LDS byte offset

    f32x8 acc[2][4];
    #pragma unroll
    for (int a = 0; a < 2; ++a)
        #pragma unroll
        for (int j = 0; j < 4; ++j)
            #pragma unroll
            for (int r = 0; r < 8; ++r) acc[a][j][r] = 0.0f;

    // ---------------- Layer 1: K = 9 patches x 128 channels ----------------
    for (int p = 0; p < 9; ++p) {
        const int di = p / 3, dj = p % 3;
        __syncthreads();   // all reads of Xs/Ws from previous patch done

        // Kick off async DMA of this patch's weights: [256 n][128 k] bf16,
        // padded to WS_STRIDE in LDS. 4096 x 16B chunks, 16 per thread.
        const char* wsrc = (const char*)w0c_ + (size_t)p * (F_ * C_ * 2);
        #pragma unroll
        for (int it = 0; it < 16; ++it) {
            const int idx = t + it * 256;        // 0..4095
            const int row = idx >> 4;            // n row 0..255
            const int c16 = idx & 15;            // 16B chunk within row
            async_ld_b128(ws_base + row * (WS_STRIDE * 2) + c16 * 16,
                          wsrc + row * 256 + c16 * 16);
        }

        // Meanwhile: fill Xs[64][128] bf16 (fp32 load + convert, coalesced).
        #pragma unroll
        for (int it = 0; it < 8; ++it) {
            const int idx = t + it * 256;
            const int row = idx >> 5;            // 0..63 within block
            const int c4  = idx & 31;            // float4 index within channels
            const int np  = n0 + (row >> 5);     // output position
            const int b   = row & 31;            // batch
            const int oh  = np / WO, ow = np % WO;
            const float4 v = *(const float4*)(
                in + ((size_t)((b * HI + oh + di) * WI + (ow + dj)) * C_) + c4 * 4);
            const unsigned int lo = (unsigned)f2bf(v.x) | ((unsigned)f2bf(v.y) << 16);
            const unsigned int hi = (unsigned)f2bf(v.z) | ((unsigned)f2bf(v.w) << 16);
            *(uint2*)((unsigned short*)Xs + row * XS_STRIDE + c4 * 4) =
                make_uint2(lo, hi);
        }
        wait_async0();       // this wave's DMA chunks landed in LDS
        __syncthreads();     // everyone's Xs stores + DMAs visible

        #pragma unroll
        for (int ks = 0; ks < 4; ++ks) {         // 4 x K32 per 128-ch patch
            Frag A[2];
            #pragma unroll
            for (int a = 0; a < 2; ++a) {
                const __bf16* ab = Xs + ((mi2 * 2 + a) * 16 + lrow) * XS_STRIDE
                                      + ks * 32 + lhalf * 8;
                A[a].u[0] = *(const uint4v*)(ab);        // K[0..7]   per half
                A[a].u[1] = *(const uint4v*)(ab + 16);   // K[16..23] per half
            }
            #pragma unroll
            for (int j = 0; j < 4; ++j) {
                const int n = (q * 4 + j) * 16 + lrow;
                Frag Bf;
                const __bf16* bb = Ws + n * WS_STRIDE + ks * 32 + lhalf * 8;
                Bf.u[0] = *(const uint4v*)(bb);
                Bf.u[1] = *(const uint4v*)(bb + 16);
                #pragma unroll
                for (int a = 0; a < 2; ++a)
                    acc[a][j] = __builtin_amdgcn_wmma_f32_16x16x32_bf16(
                        false, A[a].v, false, Bf.v, (short)0, acc[a][j],
                        false, false);
            }
        }
    }

    // bias + ReLU -> H1s (bf16)
    #pragma unroll
    for (int j = 0; j < 4; ++j) {
        const int n = (q * 4 + j) * 16 + lrow;
        const float bias = b0[n];
        #pragma unroll
        for (int a = 0; a < 2; ++a)
            #pragma unroll
            for (int r = 0; r < 8; ++r) {
                float v = acc[a][j][r] + bias;
                v = v > 0.0f ? v : 0.0f;
                H1su[((mi2 * 2 + a) * 16 + lhalf * 8 + r) * H1_STRIDE + n] =
                    f2bf(v);
            }
    }
    __syncthreads();   // H1 complete; Ws region now free for H2s

    // ---------------- Layer 2: K = 256 ----------------
    #pragma unroll
    for (int a = 0; a < 2; ++a)
        #pragma unroll
        for (int j = 0; j < 4; ++j)
            #pragma unroll
            for (int r = 0; r < 8; ++r) acc[a][j][r] = 0.0f;

    #pragma unroll
    for (int ks = 0; ks < 8; ++ks) {
        Frag A[2];
        #pragma unroll
        for (int a = 0; a < 2; ++a) {
            const __bf16* ab = H1s + ((mi2 * 2 + a) * 16 + lrow) * H1_STRIDE
                                   + ks * 32 + lhalf * 8;
            A[a].u[0] = *(const uint4v*)(ab);
            A[a].u[1] = *(const uint4v*)(ab + 16);
        }
        #pragma unroll
        for (int j = 0; j < 4; ++j) {
            const int n = (q * 4 + j) * 16 + lrow;
            Frag Bf;
            const __bf16* bb = w1t + n * F_ + ks * 32 + lhalf * 8;
            Bf.u[0] = *(const uint4v*)(bb);
            Bf.u[1] = *(const uint4v*)(bb + 16);
            #pragma unroll
            for (int a = 0; a < 2; ++a)
                acc[a][j] = __builtin_amdgcn_wmma_f32_16x16x32_bf16(
                    false, A[a].v, false, Bf.v, (short)0, acc[a][j],
                    false, false);
        }
    }

    // bias + ReLU -> H2s staged transposed: H2s[f][row]
    #pragma unroll
    for (int j = 0; j < 4; ++j) {
        const int n = (q * 4 + j) * 16 + lrow;          // output feature f
        const float bias = b1[n];
        #pragma unroll
        for (int a = 0; a < 2; ++a)
            #pragma unroll
            for (int r = 0; r < 8; ++r) {
                float v = acc[a][j][r] + bias;
                v = v > 0.0f ? v : 0.0f;
                H2s[n * H2_STRIDE + ((mi2 * 2 + a) * 16 + lhalf * 8 + r)] = v;
            }
    }
    __syncthreads();

    // Permuted write-out: out[f*(N*B) + n_pos*B + b], 128B-coalesced per wave.
    #pragma unroll
    for (int it = 0; it < 64; ++it) {
        const int idx  = t + it * 256;       // 0..16383
        const int f    = idx >> 6;
        const int sub  = idx & 63;
        const int nidx = sub >> 5;           // 0 or 1
        const int b    = sub & 31;
        const float v  = H2s[f * H2_STRIDE + (nidx * 32 + b)];
        out[(size_t)f * NB + (size_t)(n0 + nidx) * B_ + b] = v;
    }
}

// ---------------------------------------------------------------------------
extern "C" void kernel_launch(void* const* d_in, const int* in_sizes, int n_in,
                              void* d_out, int out_size, void* d_ws, size_t ws_size,
                              hipStream_t stream) {
    const float* in = (const float*)d_in[0];   // [32,64,64,128]
    const float* w0 = (const float*)d_in[1];   // [1152,256]
    const float* b0 = (const float*)d_in[2];   // [256]
    const float* w1 = (const float*)d_in[3];   // [256,256]
    const float* b1 = (const float*)d_in[4];   // [256]

    unsigned short* w0c = (unsigned short*)d_ws;            // 294912 elems
    unsigned short* w1t = w0c + 9 * F_ * C_;                // 65536 elems

    const int prep_total = 9 * F_ * C_ + F_ * F_;           // 360448
    prep_weights<<<(prep_total + 255) / 256, 256, 0, stream>>>(w0, w1, w0c, w1t);

    mlpconv_fused<<<NPOS / 2, 256, (size_t)SMEM_SZ, stream>>>(
        in, w0c, b0, w1t, b1, (float*)d_out);
}
